// GRU_77506979824073
// MI455X (gfx1250) — matrix operand order repaired and checked by reading, not compiled, not run on hardware
//
#include <hip/hip_runtime.h>

typedef __attribute__((ext_vector_type(16))) _Float16 v16h;
typedef __attribute__((ext_vector_type(8)))  _Float16 v8h;
typedef __attribute__((ext_vector_type(2)))  _Float16 v2h;
typedef __attribute__((ext_vector_type(8)))  float    v8f;
typedef __attribute__((ext_vector_type(2)))  float    v2f;

#define BB 512   // batch
#define TT 512   // timesteps
#define DD 128   // input features
#define HH 256   // hidden
#define OO 10    // outputs
#define WS (DD + HH)   // W_rz row stride = 384
#define BQ 16    // batch rows per block
#define SP (DD + 8)    // spike LDS row stride (pad -> conflict-free b128)
#define HP (HH + 8)    // hidden LDS row stride
#define NTHREADS 512   // 16 wave32s
#define PF_AHEAD 8     // L2 prefetch distance (timesteps)

__global__ __launch_bounds__(NTHREADS, 1)
void snn_gru_persistent(const float* __restrict__ x,
                        const float* __restrict__ W_rz,
                        const float* __restrict__ W_ho,
                        float* __restrict__ out)
{
    __shared__ __align__(16) _Float16 s_lds[BQ * SP];  // spike tile (f16 0/1)
    __shared__ __align__(16) _Float16 h_lds[BQ * HP];  // hidden state (f16)

    const int tid  = threadIdx.x;
    const int wv   = tid >> 5;       // wave id 0..15 -> N tile of H
    const int lane = tid & 31;
    const int half = lane >> 4;      // lane half per WMMA layout
    const int ln   = lane & 15;
    const int n0   = wv * 16;        // this wave's H columns [n0, n0+16)
    const int b0   = blockIdx.x * BQ;

    // ---- persistent B fragments (weights) in VGPRs, f32 -> f16 ----
    // B 32x16 layout: lanes 0-15 hold K=0..7,16..23, lanes 16-31 hold K=8..15,24..31,
    // column = n0+ln.  Wx[k][n] = W_rz[n*WS + k]; Wh[k][n] = W_rz[n*WS + DD + k]
    // -> the 16 K-values per lane are two contiguous runs of 8 in memory.
    v16h bx[4];   // Wx: D=128 -> 4 k-blocks of 32
    v16h bh[8];   // Wh: H=256 -> 8 k-blocks of 32
    {
        const float* wrow = W_rz + (size_t)(n0 + ln) * WS;
        #pragma unroll
        for (int kb = 0; kb < 4; ++kb) {
            const float* p = wrow + kb * 32 + half * 8;
            #pragma unroll
            for (int j = 0; j < 16; ++j) bx[kb][j] = (_Float16)p[j + (j >= 8 ? 8 : 0)];
        }
        #pragma unroll
        for (int kb = 0; kb < 8; ++kb) {
            const float* p = wrow + DD + kb * 32 + half * 8;
            #pragma unroll
            for (int j = 0; j < 16; ++j) bh[kb][j] = (_Float16)p[j + (j >= 8 ? 8 : 0)];
        }
    }

    // ---- h0 = 0 ----
    for (int i = tid; i < BQ * HP; i += NTHREADS) h_lds[i] = (_Float16)0.0f;

    // x tile per step = 16 rows x 128 f32 = 1024 float2; 2 float2 per thread.
    // pair index -> row = idx >> 6, d = (idx & 63) * 2  (coalesced b64 loads)
    // ---- prefetch x tile for t = 0 (streamed once -> non-temporal) ----
    v2f px[2];
    #pragma unroll
    for (int i = 0; i < 2; ++i) {
        int idx = tid + i * NTHREADS;
        int row = idx >> 6, d = (idx & 63) * 2;
        px[i] = __builtin_nontemporal_load(
            (const v2f*)(x + ((size_t)(b0 + row) * TT + 0) * DD + d));
    }

    __syncthreads();

    for (int t = 0; t < TT; ++t) {
        // quantized-LIF spike encode -> LDS, packed 2xf16 per ds_store_b32
        // (the clip in the reference is irrelevant to the >0.875 threshold)
        #pragma unroll
        for (int i = 0; i < 2; ++i) {
            int idx = tid + i * NTHREADS;
            int row = idx >> 6, d = (idx & 63) * 2;
            float m0 = rintf(px[i][0] * 16384.0f) * (1.0f / 16384.0f);
            float m1 = rintf(px[i][1] * 16384.0f) * (1.0f / 16384.0f);
            v2h pk;
            pk[0] = (m0 > 0.875f) ? (_Float16)1.0f : (_Float16)0.0f;
            pk[1] = (m1 > 0.875f) ? (_Float16)1.0f : (_Float16)0.0f;
            *(v2h*)(s_lds + row * SP + d) = pk;
        }
        __syncthreads();   // spikes + previous h stores visible

        // register prefetch of t+1 (hides load latency behind the WMMA chain)
        if (t + 1 < TT) {
            #pragma unroll
            for (int i = 0; i < 2; ++i) {
                int idx = tid + i * NTHREADS;
                int row = idx >> 6, d = (idx & 63) * 2;
                px[i] = __builtin_nontemporal_load(
                    (const v2f*)(x + ((size_t)(b0 + row) * TT + (t + 1)) * DD + d));
            }
        }
        // deep L2 prefetch so the HBM stream stays ahead of the recurrence
        if (t + PF_AHEAD < TT) {
            #pragma unroll
            for (int i = 0; i < 2; ++i) {
                int idx = tid + i * NTHREADS;
                int row = idx >> 6, d = (idx & 63) * 2;
                __builtin_prefetch(
                    x + ((size_t)(b0 + row) * TT + (t + PF_AHEAD)) * DD + d, 0, 1);
            }
        }

        // acc = S @ Wx + h @ Wh : two independent accumulator chains (6 deep
        // each instead of 12) to cut the serial WMMA dependency chain.
        v8f acc0 = {};
        v8f acc1 = {};

        // A 16x32 f16 layout: lane row M=ln; half 0: K kb*32+[0..7], +[16..23];
        // half 1: K kb*32+[8..15], +[24..31] -> two contiguous b128 LDS loads.
        const _Float16* srow = s_lds + ln * SP + half * 8;
        const _Float16* hrow = h_lds + ln * HP + half * 8;

        #pragma unroll
        for (int kb = 0; kb < 4; ++kb) {          // S @ Wx  -> chain 0
            const _Float16* rp = srow + kb * 32;
            v8h lo = *(const v8h*)rp;
            v8h hi = *(const v8h*)(rp + 16);
            v16h a;
            #pragma unroll
            for (int j = 0; j < 8; ++j) { a[j] = lo[j]; a[8 + j] = hi[j]; }
            acc0 = __builtin_amdgcn_wmma_f32_16x16x32_f16(
                false, a, false, bx[kb], (short)0, acc0, false, false);
        }
        #pragma unroll
        for (int kb = 0; kb < 8; ++kb) {          // h @ Wh  -> alternate chains
            const _Float16* rp = hrow + kb * 32;
            v8h lo = *(const v8h*)rp;
            v8h hi = *(const v8h*)(rp + 16);
            v16h a;
            #pragma unroll
            for (int j = 0; j < 8; ++j) { a[j] = lo[j]; a[8 + j] = hi[j]; }
            if (kb < 6)
                acc1 = __builtin_amdgcn_wmma_f32_16x16x32_f16(
                    false, a, false, bh[kb], (short)0, acc1, false, false);
            else
                acc0 = __builtin_amdgcn_wmma_f32_16x16x32_f16(
                    false, a, false, bh[kb], (short)0, acc0, false, false);
        }

        __syncthreads();   // all h/s reads done before overwriting h

        // h_new = sigmoid(acc0+acc1); C/D layout: VGPR r -> M = r+8*half, N=ln
        #pragma unroll
        for (int r = 0; r < 8; ++r) {
            int m = r + half * 8;
            float v = acc0[r] + acc1[r];
            float s = 1.0f / (1.0f + __expf(-v));
            h_lds[m * HP + n0 + ln] = (_Float16)s;
        }
        // next iteration's first barrier orders these stores before reads
    }

    __syncthreads();

    // ---- out = h_fin @ W_ho^T  ([16 x 256] x [256 x 10], tiny) ----
    if (tid < BQ * OO) {
        int b = tid / OO, o = tid % OO;
        const float* wo = W_ho + o * HH;
        float accf = 0.0f;
        #pragma unroll 8
        for (int k = 0; k < HH; ++k)
            accf += (float)h_lds[b * HP + k] * wo[k];
        out[(size_t)(b0 + b) * OO + o] = accf;
    }
}

extern "C" void kernel_launch(void* const* d_in, const int* in_sizes, int n_in,
                              void* d_out, int out_size, void* d_ws, size_t ws_size,
                              hipStream_t stream) {
    const float* x    = (const float*)d_in[0];   // [512, 512, 128] f32
    const float* W_rz = (const float*)d_in[1];   // [256, 384] f32
    const float* W_ho = (const float*)d_in[2];   // [10, 256] f32
    float* out = (float*)d_out;                  // [512, 10] f32

    dim3 grid(BB / BQ);      // 32 persistent blocks
    dim3 block(NTHREADS);    // 16 wave32s: one N-tile of H per wave
    snn_gru_persistent<<<grid, block, 0, stream>>>(x, W_rz, W_ho, out);
}
